// MultiheadAttention_37666863186394
// MI455X (gfx1250) — matrix operand order repaired and checked by reading, not compiled, not run on hardware
//
#include <hip/hip_runtime.h>

typedef __attribute__((ext_vector_type(16))) _Float16 v16h;
typedef __attribute__((ext_vector_type(8)))  float    v8f;
typedef __attribute__((ext_vector_type(4)))  unsigned v4u;
typedef __attribute__((ext_vector_type(8)))  int      v8i;
typedef __attribute__((ext_vector_type(4)))  int      v4i;

union FragU { v16h v; uint4 q[2]; };
union Pack4 { _Float16 h[4]; uint2 u2; };
union Pack8 { _Float16 h[8]; uint4 u4; };

#define WMMA_F16(a, b, c) \
    __builtin_amdgcn_wmma_f32_16x16x32_f16(false, (a), false, (b), (short)0, (c), false, false)

static constexpr int NB = 4;         // batch
static constexpr int SS = 2048;      // sequence
static constexpr int EE = 1024;      // embed
static constexpr int HH = 16;        // heads
static constexpr int DH = 64;        // head dim
static constexpr int MTOT = NB * SS; // 8192
static constexpr int LDK = 40;       // LDS row stride in halves (80 B = 5*16 B)

// ---------------------------------------------------------------------------
// TDM: 2D tile load Global -> LDS (D# per CDNA5 ISA ch.8).
// dsz: 0=1B,1=2B,2=4B. Dims/strides in elements. Rows land contiguous in LDS.
// ---------------------------------------------------------------------------
__device__ __forceinline__ unsigned lds_off(const void* p)
{
    return (unsigned)(uintptr_t)p;   // low 32 bits of generic LDS addr = LDS offset
}

__device__ __forceinline__ void tdm_load_2d(unsigned ldsAddr, const void* gptr,
                                            unsigned dsz, unsigned tdim0, unsigned tdim1,
                                            unsigned tstride0, unsigned tile0, unsigned tile1)
{
    unsigned long long ga = (unsigned long long)(uintptr_t)gptr;
    v4u g0 = {};
    g0[0] = 1u;                                               // count=1, user desc
    g0[1] = ldsAddr;                                          // [63:32] lds_addr
    g0[2] = (unsigned)ga;                                     // global_addr[31:0]
    g0[3] = (unsigned)((ga >> 32) & 0x01FFFFFFu) | (2u << 30);// addr[56:32] | type=2
    v8i g1 = {};
    g1[0] = (int)(dsz << 16);                                 // data_size; wg_mask=0
    g1[1] = (int)((tdim0 & 0xFFFFu) << 16);                   // tensor_dim0 lo16
    g1[2] = (int)(((tdim0 >> 16) & 0xFFFFu) | ((tdim1 & 0xFFFFu) << 16));
    g1[3] = (int)(((tdim1 >> 16) & 0xFFFFu) | ((tile0 & 0xFFFFu) << 16));
    g1[4] = (int)(tile1 & 0xFFFFu);                           // tile_dim1; tile_dim2=0
    g1[5] = (int)tstride0;                                    // tensor_dim0_stride lo32
    g1[6] = 0;
    g1[7] = 0;
    v4i gz4 = {};
    v8i gz8 = {};
    __builtin_amdgcn_tensor_load_to_lds(g0, g1, gz4, gz4, gz8, 0);
}

// ---------------------------------------------------------------------------
// GEMM: Out[M,N] = A[M,K] @ W[N,K]^T + bias   (M=8192, N=K=1024)
// MODE 0: out f16, per-head layout  [n,h,s,d]   (Q and K projections)
// MODE 1: out f16, transposed head  [n,h,d,s]   (V projection -> Vt)
// MODE 2: A is f16 (ctx), out f32 row-major [m,e] (final projection)
// ---------------------------------------------------------------------------
template <int MODE>
__global__ __launch_bounds__(256)
void gemm_xwT_kernel(const void* __restrict__ Aptr, const float* __restrict__ W,
                     const float* __restrict__ bias, void* __restrict__ Out)
{
    __shared__ __align__(16) _Float16 As[128 * LDK];
    __shared__ __align__(16) _Float16 Bs[128 * LDK];

    constexpr bool aF16 = (MODE == 2);
    const int tid  = threadIdx.x;
    const int lane = tid & 31;
    const int w    = tid >> 5;
    const int m0   = blockIdx.y * 128;
    const int n0   = blockIdx.x * 128;
    const int wm   = (w & 3) * 32;   // wave covers 32 rows (2 M-tiles)
    const int wn   = (w >> 2) * 64;  // and 64 cols (4 N-tiles)
    const int li   = lane & 15;
    const int half = lane >> 4;
    const int hb   = half * 8;       // A-frag K half-base
    const int kb   = half * 16;      // unused; B-frag also uses hb

    const v8f vzero = {};
    v8f acc[2][4];
#pragma unroll
    for (int t = 0; t < 2; ++t)
#pragma unroll
        for (int u = 0; u < 4; ++u) acc[t][u] = vzero;

    for (int kk = 0; kk < EE; kk += 32) {
        // -------- stage A tile (128 x 32) into LDS as f16 --------
        if constexpr (aF16) {
            const _Float16* A16 = (const _Float16*)Aptr;
#pragma unroll
            for (int i = 0; i < 2; ++i) {
                int c = tid + 256 * i, row = c >> 2, ch = c & 3;
                uint4 v = *(const uint4*)&A16[(size_t)(m0 + row) * EE + kk + ch * 8];
                *(uint4*)&As[row * LDK + ch * 8] = v;
            }
        } else {
            const float* A32 = (const float*)Aptr;
#pragma unroll
            for (int i = 0; i < 4; ++i) {
                int c = tid + 256 * i, row = c >> 3, ch = c & 7;
                float4 f = *(const float4*)&A32[(size_t)(m0 + row) * EE + kk + ch * 4];
                Pack4 p;
                p.h[0] = (_Float16)f.x; p.h[1] = (_Float16)f.y;
                p.h[2] = (_Float16)f.z; p.h[3] = (_Float16)f.w;
                *(uint2*)&As[row * LDK + ch * 4] = p.u2;
            }
        }
        // -------- stage B tile: rows = output features n, cols = k --------
#pragma unroll
        for (int i = 0; i < 4; ++i) {
            int c = tid + 256 * i, row = c >> 3, ch = c & 7;
            float4 f = *(const float4*)&W[(size_t)(n0 + row) * EE + kk + ch * 4];
            Pack4 p;
            p.h[0] = (_Float16)f.x; p.h[1] = (_Float16)f.y;
            p.h[2] = (_Float16)f.z; p.h[3] = (_Float16)f.w;
            *(uint2*)&Bs[row * LDK + ch * 4] = p.u2;
        }
        // -------- prefetch next k-tile into cache while this one computes ---
        if (kk + 32 < EE) {
            int r = tid & 127;
            if (tid < 128) {
                if constexpr (aF16)
                    __builtin_prefetch(&((const _Float16*)Aptr)[(size_t)(m0 + r) * EE + kk + 32], 0, 1);
                else
                    __builtin_prefetch(&((const float*)Aptr)[(size_t)(m0 + r) * EE + kk + 32], 0, 1);
            } else {
                __builtin_prefetch(&W[(size_t)(n0 + r) * EE + kk + 32], 0, 1);
            }
        }
        __syncthreads();

        FragU fa[2], fb[4];
#pragma unroll
        for (int t = 0; t < 2; ++t) {
            int row = wm + t * 16 + li;
            fa[t].q[0] = *(const uint4*)&As[row * LDK + hb];
            fa[t].q[1] = *(const uint4*)&As[row * LDK + 16 + hb];
        }
#pragma unroll
        for (int u = 0; u < 4; ++u) {
            int row = wn + u * 16 + li;
            fb[u].q[0] = *(const uint4*)&Bs[row * LDK + hb];
            fb[u].q[1] = *(const uint4*)&Bs[row * LDK + 16 + hb];
        }
#pragma unroll
        for (int t = 0; t < 2; ++t)
#pragma unroll
            for (int u = 0; u < 4; ++u)
                acc[t][u] = WMMA_F16(fa[t].v, fb[u].v, acc[t][u]);
        __syncthreads();
    }

    // -------- epilogue: bias + store --------
#pragma unroll
    for (int t = 0; t < 2; ++t) {
        const int mbase = m0 + wm + t * 16 + half * 8; // first of 8 rows in this lane
#pragma unroll
        for (int u = 0; u < 4; ++u) {
            const int ncol = n0 + wn + u * 16 + li;
            const float bv = bias[ncol];
            const int h = ncol >> 6, d = ncol & 63;
            if constexpr (MODE == 1) {
                // Vt[n,h,d,s]: the 8 rows (s) are contiguous -> one b128 store
                const int batch = mbase >> 11, s = mbase & (SS - 1);
                Pack8 p;
#pragma unroll
                for (int i = 0; i < 8; ++i) p.h[i] = (_Float16)(acc[t][u][i] + bv);
                _Float16* o = (_Float16*)Out;
                *(uint4*)&o[(((size_t)batch * HH + h) * DH + d) * SS + s] = p.u4;
            } else if constexpr (MODE == 0) {
                _Float16* o = (_Float16*)Out;
#pragma unroll
                for (int i = 0; i < 8; ++i) {
                    int mrow = mbase + i, batch = mrow >> 11, s = mrow & (SS - 1);
                    o[(((size_t)batch * HH + h) * SS + s) * DH + d] =
                        (_Float16)(acc[t][u][i] + bv);
                }
            } else {
                float* o = (float*)Out;
#pragma unroll
                for (int i = 0; i < 8; ++i)
                    o[(size_t)(mbase + i) * EE + ncol] = acc[t][u][i] + bv;
            }
        }
    }
}

// ---------------------------------------------------------------------------
// scores + softmax: one block owns 16 query rows of one (n,h); the full
// 16x2048 score strip lives in registers (8 waves x 16 col-tiles x v8f).
// K tiles arrive via wave-private double-buffered TDM (linear 2KB copies).
// ---------------------------------------------------------------------------
__global__ __launch_bounds__(256)
void attn_scores_softmax_kernel(const _Float16* __restrict__ Q,
                                const _Float16* __restrict__ K,
                                float* __restrict__ attn)
{
    __shared__ __align__(16) _Float16 Kbuf[8][2][16 * DH]; // 32 KB, wave-private
    __shared__ float wred[8 * 16];

    const int tid  = threadIdx.x;
    const int lane = tid & 31;
    const int w    = tid >> 5;
    const int nh   = blockIdx.x >> 7;
    const int s0   = (blockIdx.x & 127) * 16;
    const int li   = lane & 15;
    const int half = lane >> 4;
    const float scale = 0.125f; // 1/sqrt(64)

    // Q fragments (16 rows x 64 K) -- two 16x32 A-frags, all b128 loads
    const int hb = half * 8;
    const _Float16* qp = Q + ((size_t)nh * SS + s0 + li) * DH;
    FragU a0, a1;
    a0.q[0] = *(const uint4*)&qp[hb];      a0.q[1] = *(const uint4*)&qp[16 + hb];
    a1.q[0] = *(const uint4*)&qp[32 + hb]; a1.q[1] = *(const uint4*)&qp[48 + hb];

    const int kb = half * 16;
    const _Float16* kbase = K + (size_t)nh * SS * DH;
    const v8f vzero = {};
    v8f acc[16];

    // K tile for this wave's columns: 16 rows x 64 halves, fully contiguous
    // -> describe as a 1-row 2KB linear TDM copy.
    tdm_load_2d(lds_off(&Kbuf[w][0][0]), kbase + (size_t)(w * 256) * DH,
                1u, 1024u, 1u, 1024u, 1024u, 1u);
    for (int ct = 0; ct < 16; ++ct) {
        if (ct + 1 < 16) {
            tdm_load_2d(lds_off(&Kbuf[w][(ct + 1) & 1][0]),
                        kbase + (size_t)(w * 256 + (ct + 1) * 16) * DH,
                        1u, 1024u, 1u, 1024u, 1024u, 1u);
            __builtin_amdgcn_s_wait_tensorcnt(1);
        } else {
            __builtin_amdgcn_s_wait_tensorcnt(0);
        }
        const _Float16* kl = &Kbuf[w][ct & 1][0];
        FragU b0, b1;
        b0.q[0] = *(const uint4*)&kl[li * DH + hb];
        b0.q[1] = *(const uint4*)&kl[li * DH + 16 + hb];
        b1.q[0] = *(const uint4*)&kl[li * DH + 32 + hb];
        b1.q[1] = *(const uint4*)&kl[li * DH + 48 + hb];
        v8f c = vzero;
        c = WMMA_F16(a0.v, b0.v, c);
        c = WMMA_F16(a1.v, b1.v, c);
        acc[ct] = c;
    }
#pragma unroll
    for (int ct = 0; ct < 16; ++ct)
#pragma unroll
        for (int i = 0; i < 8; ++i) acc[ct][i] *= scale;

    // ---- row max: per-lane over 16 tiles, then shfl across the 16-lane half
    float pm[8];
#pragma unroll
    for (int i = 0; i < 8; ++i) {
        float m = acc[0][i];
#pragma unroll
        for (int ct = 1; ct < 16; ++ct) m = fmaxf(m, acc[ct][i]);
        pm[i] = m;
    }
#pragma unroll
    for (int off = 1; off < 16; off <<= 1)
#pragma unroll
        for (int i = 0; i < 8; ++i) pm[i] = fmaxf(pm[i], __shfl_xor(pm[i], off, 32));
    if (li == 0)
#pragma unroll
        for (int i = 0; i < 8; ++i) wred[w * 16 + half * 8 + i] = pm[i];
    __syncthreads();
    float gm[8];
#pragma unroll
    for (int i = 0; i < 8; ++i) {
        const int r = half * 8 + i;
        float m = wred[r];
        for (int w2 = 1; w2 < 8; ++w2) m = fmaxf(m, wred[w2 * 16 + r]);
        gm[i] = m;
    }
    __syncthreads();

    // ---- exp + row sum
    float ps[8];
#pragma unroll
    for (int i = 0; i < 8; ++i) ps[i] = 0.f;
#pragma unroll
    for (int ct = 0; ct < 16; ++ct)
#pragma unroll
        for (int i = 0; i < 8; ++i) {
            float e = __expf(acc[ct][i] - gm[i]);
            acc[ct][i] = e;
            ps[i] += e;
        }
#pragma unroll
    for (int off = 1; off < 16; off <<= 1)
#pragma unroll
        for (int i = 0; i < 8; ++i) ps[i] += __shfl_xor(ps[i], off, 32);
    if (li == 0)
#pragma unroll
        for (int i = 0; i < 8; ++i) wred[w * 16 + half * 8 + i] = ps[i];
    __syncthreads();
    float ginv[8];
#pragma unroll
    for (int i = 0; i < 8; ++i) {
        const int r = half * 8 + i;
        float sm = 0.f;
        for (int w2 = 0; w2 < 8; ++w2) sm += wred[w2 * 16 + r];
        ginv[i] = 1.0f / sm;
    }

    // ---- write normalized probs (coalesced 16-lane b32 stores)
    float* ob = attn + ((size_t)nh * SS + s0) * SS;
#pragma unroll
    for (int ct = 0; ct < 16; ++ct) {
        const int col = w * 256 + ct * 16 + li;
#pragma unroll
        for (int i = 0; i < 8; ++i) {
            const int r = half * 8 + i;
            ob[(size_t)r * SS + col] = acc[ct][i] * ginv[i];
        }
    }
}

// ---------------------------------------------------------------------------
// AV: ctx[16,64] = P[16,2048] @ V[2048,64]. Per-wave TDM double-buffers the
// 16x32 f32 P tile (fixes 8KB-stride lane access); wave 1 TDM-stages the
// shared 64x32 Vt slice for the whole block.
// ---------------------------------------------------------------------------
__global__ __launch_bounds__(256)
void attn_av_kernel(const float* __restrict__ Pfull, const _Float16* __restrict__ Vt,
                    _Float16* __restrict__ Ctx)
{
    __shared__ __align__(16) float    Pbuf[8][2][16 * 32]; // 32 KB, wave-private
    __shared__ __align__(16) _Float16 Vbuf[2][DH * 32];    // 8 KB, block-shared

    const int tid  = threadIdx.x;
    const int lane = tid & 31;
    const int w    = tid >> 5;
    const int nh   = blockIdx.x >> 4;
    const int s0   = (blockIdx.x & 15) * 128 + w * 16;
    const int batch = nh >> 4, h = nh & 15;
    const int li   = lane & 15;
    const int half = lane >> 4;

    const v8f vzero = {};
    v8f acc[4];
#pragma unroll
    for (int u = 0; u < 4; ++u) acc[u] = vzero;

    const float*    pbase = Pfull + ((size_t)nh * SS + s0) * SS;
    const _Float16* vbase = Vt + (size_t)nh * DH * SS;

    // prologue: stage k-step 0
    tdm_load_2d(lds_off(&Pbuf[w][0][0]), pbase, 2u, (unsigned)SS, 16u, (unsigned)SS, 32u, 16u);
    if (w == 1)
        tdm_load_2d(lds_off(&Vbuf[0][0]), vbase, 1u, (unsigned)SS, (unsigned)DH,
                    (unsigned)SS, 32u, (unsigned)DH);

    for (int ks = 0; ks < 64; ++ks) {
        if (ks + 1 < 64) {
            tdm_load_2d(lds_off(&Pbuf[w][(ks + 1) & 1][0]), pbase + (ks + 1) * 32,
                        2u, (unsigned)SS, 16u, (unsigned)SS, 32u, 16u);
            if (w == 1)
                tdm_load_2d(lds_off(&Vbuf[(ks + 1) & 1][0]), vbase + (ks + 1) * 32,
                            1u, (unsigned)SS, (unsigned)DH, (unsigned)SS, 32u, (unsigned)DH);
            if (w == 1) __builtin_amdgcn_s_wait_tensorcnt(2);
            else        __builtin_amdgcn_s_wait_tensorcnt(1);
        } else {
            __builtin_amdgcn_s_wait_tensorcnt(0);
        }
        __syncthreads(); // Vbuf[ks&1] complete (wave 1 waited) and visible

        const float* pl = &Pbuf[w][ks & 1][0];
        const int rb = li * 32 + half * 8;
        float4 f0 = *(const float4*)&pl[rb];
        float4 f1 = *(const float4*)&pl[rb + 4];
        float4 f2 = *(const float4*)&pl[rb + 16];
        float4 f3 = *(const float4*)&pl[rb + 20];
        FragU a;
        a.v[0]  = (_Float16)f0.x; a.v[1]  = (_Float16)f0.y;
        a.v[2]  = (_Float16)f0.z; a.v[3]  = (_Float16)f0.w;
        a.v[4]  = (_Float16)f1.x; a.v[5]  = (_Float16)f1.y;
        a.v[6]  = (_Float16)f1.z; a.v[7]  = (_Float16)f1.w;
        a.v[8]  = (_Float16)f2.x; a.v[9]  = (_Float16)f2.y;
        a.v[10] = (_Float16)f2.z; a.v[11] = (_Float16)f2.w;
        a.v[12] = (_Float16)f3.x; a.v[13] = (_Float16)f3.y;
        a.v[14] = (_Float16)f3.z; a.v[15] = (_Float16)f3.w;

        const _Float16* vl = &Vbuf[ks & 1][0];
#pragma unroll
        for (int u = 0; u < 4; ++u) {
            const int d = u * 16 + li;
            FragU b;
            b.q[0] = *(const uint4*)&vl[d * 32 + half * 8];
            b.q[1] = *(const uint4*)&vl[d * 32 + 16 + half * 8];
            acc[u] = WMMA_F16(a.v, b.v, acc[u]);
        }
        __syncthreads(); // all waves done reading before next iter's TDM overwrites
    }

#pragma unroll
    for (int u = 0; u < 4; ++u) {
        const int d = u * 16 + li;
#pragma unroll
        for (int i = 0; i < 8; ++i) {
            const int s = s0 + half * 8 + i;
            Ctx[((size_t)batch * SS + s) * EE + h * DH + d] = (_Float16)acc[u][i];
        }
    }
}

// ---------------------------------------------------------------------------
extern "C" void kernel_launch(void* const* d_in, const int* in_sizes, int n_in,
                              void* d_out, int out_size, void* d_ws, size_t ws_size,
                              hipStream_t stream)
{
    (void)in_sizes; (void)n_in; (void)out_size; (void)ws_size;

    const float* query = (const float*)d_in[0];
    const float* key   = (const float*)d_in[1];
    const float* value = (const float*)d_in[2];
    const float* Wq    = (const float*)d_in[3];
    const float* bq    = (const float*)d_in[4];
    const float* Wk    = (const float*)d_in[5];
    const float* bk    = (const float*)d_in[6];
    const float* Wv    = (const float*)d_in[7];
    const float* bv    = (const float*)d_in[8];
    const float* Wo    = (const float*)d_in[9];
    const float* bo    = (const float*)d_in[10];

    float* out  = (float*)d_out;                   // [N,S,E]
    float* attn = out + (size_t)NB * SS * EE;      // [N,H,S,S]

    const size_t headElems = (size_t)NB * SS * EE; // 8,388,608 halves each
    _Float16* Qb  = (_Float16*)d_ws;
    _Float16* Kb  = Qb  + headElems;
    _Float16* Vtb = Kb  + headElems;
    _Float16* Ctx = Vtb + headElems;               // 64 MB total workspace

    dim3 blk(256);
    dim3 gGemm(EE / 128, MTOT / 128);              // 8 x 64

    gemm_xwT_kernel<0><<<gGemm, blk, 0, stream>>>(query, Wq, bq, Qb);
    gemm_xwT_kernel<0><<<gGemm, blk, 0, stream>>>(key,   Wk, bk, Kb);
    gemm_xwT_kernel<1><<<gGemm, blk, 0, stream>>>(value, Wv, bv, Vtb);
    attn_scores_softmax_kernel<<<dim3(NB * HH * (SS / 16)), blk, 0, stream>>>(Qb, Kb, attn);
    attn_av_kernel<<<dim3(NB * HH * (SS / 128)), blk, 0, stream>>>(attn, Vtb, Ctx);
    gemm_xwT_kernel<2><<<gGemm, blk, 0, stream>>>(Ctx, Wo, bo, out);
}
